// BiDAF_PrNet_63660005261485
// MI455X (gfx1250) — matrix operand order repaired and checked by reading, not compiled, not run on hardware
//
#include <hip/hip_runtime.h>
#include <hip/hip_bf16.h>
#include <math.h>

typedef __attribute__((ext_vector_type(16))) _Float16 v16h;
typedef __attribute__((ext_vector_type(8)))  float    v8f;
typedef __attribute__((ext_vector_type(4)))  float    f32x4;

#define DEV static __device__ __forceinline__

DEV float sig_(float x) { return 1.0f / (1.0f + expf(-x)); }

// ================= WMMA fragment helpers (branch-free) =====================
// A: 16x32 (MxK) f16 fragment from f32 row-major src. Requires: rows m0..m0+15
// valid, cols k0..k0+31 within padded lda (lda%4==0, 16B-aligned rows).
// Layout: lanes 0-15 row=lane, elems 0..15 -> k0..+15;
// lanes 16-31 same rows, +16 on k.
DEV v16h load_a_f32p(const float* src, int lda, int m0, int k0, int lane) {
    int m = m0 + (lane & 15);
    int kb = k0 + ((lane >> 4) ? 16 : 0);
    const float* p = src + (long)m * lda + kb;
    f32x4 x0 = *(const f32x4*)(p);
    f32x4 x1 = *(const f32x4*)(p + 4);
    f32x4 x2 = *(const f32x4*)(p + 8);
    f32x4 x3 = *(const f32x4*)(p + 12);
    v16h a;
#pragma unroll
    for (int e = 0; e < 4; ++e) {
        a[e]      = (_Float16)x0[e];
        a[4 + e]  = (_Float16)x1[e];
        a[8 + e]  = (_Float16)x2[e];
        a[12 + e] = (_Float16)x3[e];
    }
    return a;
}

// B: 32x16 (KxN) f16 fragment from pre-packed f16 weights Wp[Np][Kp]
// (B[k][n] = Wp[n][k], zero padded). One contiguous 32B read per lane.
// Layout: lane n = lane&15; lanes 16-31 hold K=16..31; elems in k order.
DEV v16h load_b_h16(const _Float16* Wp, int ldb, int k0, int n0, int lane) {
    int n = n0 + (lane & 15);
    int kb = k0 + ((lane >> 4) ? 16 : 0);
    return *(const v16h*)(Wp + (long)n * ldb + kb);
}

// D: 16x16 f32. Lane n = lane&15; elem e holds m = e (+8 for lanes>=16).
// Writes n < Nstore; zero for n in [N, Nstore).
DEV void store_d(float* C, int ldc, int m0, int n0, int Mmax, int N, int Nstore,
                 const float* bias, v8f acc, int lane) {
    int n = n0 + (lane & 15);
    int madd = (lane >> 4) ? 8 : 0;
    bool valid = n < N;
    float bv = (bias != nullptr && valid) ? bias[n] : 0.0f;
#pragma unroll
    for (int e = 0; e < 8; ++e) {
        int m = m0 + e + madd;
        if (m < Mmax && n < Nstore) C[(long)m * ldc + n] = valid ? (acc[e] + bv) : 0.0f;
    }
}

// ================= Batched WMMA GEMM: C = A * Wp^T + bias ==================
__global__ void gemm_wmma_kernel(const float* __restrict__ A, long sA, int lda,
                                 const _Float16* __restrict__ Bp, long sB, int ldb,
                                 const float* __restrict__ bias,
                                 float* __restrict__ C, long sC, int ldc,
                                 int M, int N, int Nstore, int Kp) {
    int lane = threadIdx.x & 31;
    int n0 = blockIdx.x * 16;
    int m0 = blockIdx.y * 16;
    int z  = blockIdx.z;
    const float* Ab = A + (long)z * sA;
    const _Float16* Bb = Bp + (long)z * sB;
    float* Cb = C + (long)z * sC;
    v8f acc = {};
    for (int k0 = 0; k0 < Kp; k0 += 32) {
        v16h a = load_a_f32p(Ab, lda, m0, k0, lane);
        v16h b = load_b_h16(Bb, ldb, k0, n0, lane);
        acc = __builtin_amdgcn_wmma_f32_16x16x32_f16(false, a, false, b,
                                                     (short)0, acc, false, false);
    }
    store_d(Cb, ldc, m0, n0, M, N, Nstore, bias, acc, lane);
}

// ================= Persistent GRU scan (16 batch rows / block) =============
// gx: [B,T,3H] (stride 3H). whp: packed f16 [NT*16][Hp]. h kept in LDS at
// padded width Hp (zero padding -> exact results).
__global__ void gru_scan_kernel(const float* __restrict__ gx,
                                const _Float16* __restrict__ whp,
                                const float* __restrict__ bh,
                                float* __restrict__ y,
                                int B, int T, int H, int Hp, int ld_out,
                                int col_off, int dirBack) {
    extern __shared__ float smem[];
    const int H3 = 3 * H;
    const int NT = (H3 + 15) / 16;
    const int ghw = NT * 16;
    float* hbuf  = smem;             // [16 * Hp]
    float* ghbuf = smem + 16 * Hp;   // [16 * ghw]
    int tid  = threadIdx.x;
    int lane = tid & 31;
    int wid  = tid >> 5;
    int nw   = blockDim.x >> 5;
    int m0   = blockIdx.x * 16;
    const int KT = Hp >> 5;

    for (int i = tid; i < 16 * Hp; i += blockDim.x) hbuf[i] = 0.0f;
    __syncthreads();

    for (int step = 0; step < T; ++step) {
        int t = dirBack ? (T - 1 - step) : step;
        for (int ct = wid; ct < NT; ct += nw) {
            v8f acc = {};
            for (int kt = 0; kt < KT; ++kt) {
                v16h a = load_a_f32p(hbuf, Hp, 0, kt * 32, lane);
                v16h b = load_b_h16(whp, Hp, kt * 32, ct * 16, lane);
                acc = __builtin_amdgcn_wmma_f32_16x16x32_f16(false, a, false, b,
                                                             (short)0, acc, false, false);
            }
            int n = ct * 16 + (lane & 15);
            int madd = (lane >> 4) ? 8 : 0;
#pragma unroll
            for (int e = 0; e < 8; ++e) ghbuf[(e + madd) * ghw + n] = acc[e];
        }
        __syncthreads();
        for (int i = tid; i < 16 * H; i += blockDim.x) {
            int r_ = i / H, j = i % H;
            int b = m0 + r_;
            if (b < B) {
                const float* gxr = gx + ((long)b * T + t) * H3;
                float hr = ghbuf[r_ * ghw + j]         + bh[j];
                float hz = ghbuf[r_ * ghw + H + j]     + bh[H + j];
                float hn = ghbuf[r_ * ghw + 2 * H + j] + bh[2 * H + j];
                float rg = sig_(gxr[j] + hr);
                float zg = sig_(gxr[H + j] + hz);
                float ng = tanhf(gxr[2 * H + j] + rg * hn);
                float hnew = (1.0f - zg) * ng + zg * hbuf[r_ * Hp + j];
                hbuf[r_ * Hp + j] = hnew;
                y[((long)b * T + t) * ld_out + col_off + j] = hnew;
            }
        }
        __syncthreads();
    }
}

// ================= Packing / small kernels =================================
__global__ void pack_w16(const float* __restrict__ W, _Float16* __restrict__ out,
                         int N, int K, int Kp, long total) {
    long i = (long)blockIdx.x * blockDim.x + threadIdx.x;
    if (i >= total) return;
    int n = (int)(i / Kp), k = (int)(i % Kp);
    out[i] = (_Float16)((n < N && k < K) ? W[(long)n * K + k] : 0.0f);
}

// out[b][r][kp] = U[b][r][k] zero padded (U stride rows x K)
__global__ void pack_u16(const float* __restrict__ U, _Float16* __restrict__ out,
                         int rows, int K, int Kp, long total) {
    long i = (long)blockIdx.x * blockDim.x + threadIdx.x;
    if (i >= total) return;
    long br = i / Kp;          // b*rows + r
    int k = (int)(i % Kp);
    out[i] = (_Float16)((k < K) ? U[br * K + k] : 0.0f);
}

// out[b][n][k] = U[b][k][n] (transpose pack), n zero padded to Np
__global__ void pack_ut16(const float* __restrict__ U, _Float16* __restrict__ out,
                          int N, int Np, int K, long total) {
    long i = (long)blockIdx.x * blockDim.x + threadIdx.x;
    if (i >= total) return;
    int k = (int)(i % K);
    int n = (int)((i / K) % Np);
    long b = i / ((long)Np * K);
    out[i] = (_Float16)((n < N) ? U[(b * K + k) * N + n] : 0.0f);
}

__global__ void fill_zero(float* __restrict__ p, long n) {
    long i = (long)blockIdx.x * blockDim.x + threadIdx.x;
    if (i < n) p[i] = 0.0f;
}

__global__ void gather_embed(const int* __restrict__ idx, const float* __restrict__ embed,
                             float* __restrict__ out, int D, int Dp, long total) {
    long i = (long)blockIdx.x * blockDim.x + threadIdx.x;
    if (i >= total) return;
    long r = i / Dp; int d = (int)(i % Dp);
    out[i] = (d < D) ? embed[(long)idx[r] * D + d] : 0.0f;
}

__global__ void rowdot(const float* __restrict__ X, const float* __restrict__ w,
                       float* __restrict__ out, int rows, int D) {
    int r = blockIdx.x * blockDim.x + threadIdx.x;
    if (r < rows) {
        float s = 0.0f;
        for (int d = 0; d < D; ++d) s += X[(long)r * D + d] * w[d];
        out[r] = s;
    }
}

// Hw3[r][dp] = H[r][d] * w[d], padded cols zeroed
__global__ void mul_cols_pad(const float* __restrict__ X, const float* __restrict__ w,
                             float* __restrict__ out, int D, int Dp, long total) {
    long i = (long)blockIdx.x * blockDim.x + threadIdx.x;
    if (i >= total) return;
    long r = i / Dp; int d = (int)(i % Dp);
    out[i] = (d < D) ? X[r * D + d] * w[d] : 0.0f;
}

__global__ void softmax_S(float* __restrict__ S, const float* __restrict__ hw1,
                          const float* __restrict__ uw2, float* __restrict__ Smax,
                          int T, int L) {
    __shared__ float red[64];
    int row = blockIdx.x;            // b*T + t
    int b = row / T;
    int l = threadIdx.x;
    float v = S[(long)row * L + l] + hw1[row] + uw2[b * L + l];
    red[l] = v; __syncthreads();
    for (int o = 32; o > 0; o >>= 1) { if (l < o) red[l] = fmaxf(red[l], red[l + o]); __syncthreads(); }
    float m = red[0]; __syncthreads();
    float e = expf(v - m);
    red[l] = e; __syncthreads();
    for (int o = 32; o > 0; o >>= 1) { if (l < o) red[l] += red[l + o]; __syncthreads(); }
    S[(long)row * L + l] = e / red[0];
    if (l == 0) Smax[row] = m;
}

__global__ void softmax_T(const float* __restrict__ in, float* __restrict__ out, int T) {
    __shared__ float red[256];
    int b = blockIdx.x, tid = threadIdx.x;
    float m = -1e30f;
    for (int t = tid; t < T; t += blockDim.x) m = fmaxf(m, in[b * T + t]);
    red[tid] = m; __syncthreads();
    for (int o = 128; o > 0; o >>= 1) { if (tid < o) red[tid] = fmaxf(red[tid], red[tid + o]); __syncthreads(); }
    m = red[0]; __syncthreads();
    float s = 0.0f;
    for (int t = tid; t < T; t += blockDim.x) s += expf(in[b * T + t] - m);
    red[tid] = s; __syncthreads();
    for (int o = 128; o > 0; o >>= 1) { if (tid < o) red[tid] += red[tid + o]; __syncthreads(); }
    s = red[0];
    for (int t = tid; t < T; t += blockDim.x) out[b * T + t] = expf(in[b * T + t] - m) / s;
}

__global__ void q2c_kernel(const float* __restrict__ Batt, const float* __restrict__ H,
                           float* __restrict__ q2c, int T, int D2) {
    int b = blockIdx.x, d = threadIdx.x;
    if (d < D2) {
        float s = 0.0f;
        for (int t = 0; t < T; ++t) s += Batt[b * T + t] * H[((long)b * T + t) * D2 + d];
        q2c[b * D2 + d] = s;
    }
}

__global__ void build_G(const float* __restrict__ Hm, const float* __restrict__ c2q,
                        const float* __restrict__ q2c, float* __restrict__ G,
                        int B, int T, int D2) {
    long i = (long)blockIdx.x * blockDim.x + threadIdx.x;
    long total = (long)B * T * 4 * D2;
    if (i >= total) return;
    int c = (int)(i % (4 * D2));
    long row = i / (4 * D2);
    int b = (int)(row / T);
    float h = Hm[row * D2 + (c % D2)];
    float out;
    if (c < D2)            out = h;
    else if (c < 2 * D2)   out = c2q[row * D2 + (c - D2)];
    else if (c < 3 * D2)   out = h * c2q[row * D2 + (c - 2 * D2)];
    else                   out = h * q2c[(long)b * D2 + (c - 3 * D2)];
    G[i] = out;
}

__global__ void dec_cell(const float* __restrict__ hid_in, int useZero,
                         const float* __restrict__ wh, const float* __restrict__ bi,
                         const float* __restrict__ bh, const float* __restrict__ w2,
                         float* __restrict__ hid_out, float* __restrict__ w2h,
                         int B, int Hd, int P) {
    extern __shared__ float sm[];
    float* hp = sm;
    float* hn = sm + B * Hd;
    int tid = threadIdx.x;
    for (int i = tid; i < B * Hd; i += blockDim.x) hp[i] = useZero ? 0.0f : hid_in[i];
    __syncthreads();
    for (int i = tid; i < B * Hd; i += blockDim.x) {
        int b = i / Hd, j = i % Hd;
        const float* hrow = hp + b * Hd;
        const float* wr = wh + (long)j * Hd;
        const float* wz = wh + (long)(Hd + j) * Hd;
        const float* wn = wh + (long)(2 * Hd + j) * Hd;
        float gr = 0.0f, gz = 0.0f, gn = 0.0f;
        for (int k = 0; k < Hd; ++k) {
            float hk = hrow[k];
            gr += hk * wr[k]; gz += hk * wz[k]; gn += hk * wn[k];
        }
        float r = sig_(bi[j] + gr + bh[j]);
        float z = sig_(bi[Hd + j] + gz + bh[Hd + j]);
        float n = tanhf(bi[2 * Hd + j] + r * (gn + bh[2 * Hd + j]));
        hn[i] = (1.0f - z) * n + z * hrow[j];
    }
    __syncthreads();
    for (int i = tid; i < B * Hd; i += blockDim.x) hid_out[i] = hn[i];
    for (int i = tid; i < B * P; i += blockDim.x) {
        int b = i / P, p = i % P;
        const float* hrow = hn + b * Hd;
        const float* wrow = w2 + (long)p * Hd;
        float s = 0.0f;
        for (int k = 0; k < Hd; ++k) s += hrow[k] * wrow[k];
        w2h[i] = s;
    }
}

__global__ void blend_kernel(const float* __restrict__ encW1, const float* __restrict__ w2h,
                             const float* __restrict__ vt, float* __restrict__ out,
                             int B, int T, int P) {
    int i = blockIdx.x * blockDim.x + threadIdx.x;
    if (i >= B * T) return;
    int b = i / T;
    const float* er = encW1 + (long)i * P;
    const float* wr = w2h + (long)b * P;
    float s = 0.0f;
    for (int p = 0; p < P; ++p) s += tanhf(er[p] + wr[p]) * vt[p];
    out[i] = s;
}

__global__ void logsoftmax_T(const float* __restrict__ in, float* __restrict__ out,
                             int T, int step, int nsteps) {
    __shared__ float red[256];
    int b = blockIdx.x, tid = threadIdx.x;
    float m = -1e30f;
    for (int t = tid; t < T; t += blockDim.x) m = fmaxf(m, in[b * T + t]);
    red[tid] = m; __syncthreads();
    for (int o = 128; o > 0; o >>= 1) { if (tid < o) red[tid] = fmaxf(red[tid], red[tid + o]); __syncthreads(); }
    m = red[0]; __syncthreads();
    float s = 0.0f;
    for (int t = tid; t < T; t += blockDim.x) s += expf(in[b * T + t] - m);
    red[tid] = s; __syncthreads();
    for (int o = 128; o > 0; o >>= 1) { if (tid < o) red[tid] += red[tid + o]; __syncthreads(); }
    float lse = m + logf(red[0]);
    for (int t = tid; t < T; t += blockDim.x)
        out[((long)b * nsteps + step) * T + t] = in[b * T + t] - lse;
}

// ================= Host-side orchestration =================================
static inline void gemm(hipStream_t s, const float* A, long sA, int lda,
                        const _Float16* Bp, long sB, int ldb,
                        const float* bias, float* C, long sC, int ldc,
                        int M, int N, int Nstore, int Kp, int batch) {
    dim3 g((Nstore + 15) / 16, M / 16, batch);
    gemm_wmma_kernel<<<g, 32, 0, s>>>(A, sA, lda, Bp, sB, ldb, bias, C, sC, ldc,
                                      M, N, Nstore, Kp);
}

static inline void gru_scan(hipStream_t s, const float* gx, const _Float16* whp,
                            const float* bh, float* y, int B, int T, int H, int Hp,
                            int ld_out, int col_off, int back) {
    int NT = (3 * H + 15) / 16;
    size_t shmem = (size_t)(16 * Hp + 16 * NT * 16) * sizeof(float);
    gru_scan_kernel<<<dim3((B + 15) / 16), 512, shmem, s>>>(gx, whp, bh, y, B, T, H,
                                                            Hp, ld_out, col_off, back);
}

static inline void packW(hipStream_t s, const float* W, _Float16* out,
                         int N, int K, int Np, int Kp) {
    long tot = (long)Np * Kp;
    pack_w16<<<(tot + 255) / 256, 256, 0, s>>>(W, out, N, K, Kp, tot);
}

extern "C" void kernel_launch(void* const* d_in, const int* in_sizes, int n_in,
                              void* d_out, int out_size, void* d_ws, size_t ws_size,
                              hipStream_t stream) {
    const int B = 32, T = 512, L = 64, D = 100;
    const int D2 = 200, D3 = 300, D8 = 800, Hd = 400, G12 = 1200;
    const int Dp = 128, D2p = 224, Hdp = 416;        // padded K dims (mult 32)
    const int D3p16 = 304;                           // 300 -> tile-padded rows
    const long BT = (long)B * T;
    const long BL = (long)B * L;

    const int*   x_word  = (const int*)d_in[0];
    const int*   x_query = (const int*)d_in[1];
    const float* embed   = (const float*)d_in[2];
    const float* ctx_wi  = (const float*)d_in[3];
    const float* ctx_wh  = (const float*)d_in[4];
    const float* ctx_bi  = (const float*)d_in[5];
    const float* ctx_bh  = (const float*)d_in[6];
    const float* mod0_wi = (const float*)d_in[7];
    const float* mod0_wh = (const float*)d_in[8];
    const float* mod0_bi = (const float*)d_in[9];
    const float* mod0_bh = (const float*)d_in[10];
    const float* mod1_wi = (const float*)d_in[11];
    const float* mod1_wh = (const float*)d_in[12];
    const float* mod1_bi = (const float*)d_in[13];
    const float* mod1_bh = (const float*)d_in[14];
    const float* w_sim   = (const float*)d_in[15];
    const float* enc_wi  = (const float*)d_in[16];
    const float* enc_wh  = (const float*)d_in[17];
    const float* enc_bi  = (const float*)d_in[18];
    const float* enc_bh  = (const float*)d_in[19];
    /* dec_wi unused: dec_in is always zero */
    const float* dec_wh  = (const float*)d_in[21];
    const float* dec_bi  = (const float*)d_in[22];
    const float* dec_bh  = (const float*)d_in[23];
    const float* w1      = (const float*)d_in[24];
    const float* w2      = (const float*)d_in[25];
    const float* vt      = (const float*)d_in[26];
    (void)in_sizes; (void)n_in; (void)out_size; (void)ws_size;

    char* base = (char*)d_ws;
    size_t off = 0;
    auto takeF = [&](size_t n) { off = (off + 255) & ~(size_t)255; float* p = (float*)(base + off); off += n * 4; return p; };
    auto takeH = [&](size_t n) { off = (off + 255) & ~(size_t)255; _Float16* p = (_Float16*)(base + off); off += n * 2; return p; };

    // activations (padded strides where used as GEMM A)
    float* embC  = takeF(BT * Dp);
    float* embQ  = takeF(BL * Dp);
    float* Hbuf  = takeF(BT * D2);
    float* Ubuf  = takeF(BL * D2);
    float* Hw3   = takeF(BT * D2p);
    float* Sbuf  = takeF(BT * L);
    float* hw1   = takeF(BT);
    float* uw2   = takeF(BL);
    float* SmaxB = takeF(BT);
    float* Batt  = takeF(BT);
    float* q2cB  = takeF((size_t)B * D2);
    float* c2qB  = takeF(BT * D2);
    float* Gbuf  = takeF(BT * D8);
    float* GX    = takeF(BT * G12);
    float* M0    = takeF(BT * D2p);
    float* M1    = takeF(BT * D2p);
    float* encB  = takeF(BT * Hdp);
    float* encW1 = takeF(BT * D2);
    float* hid   = takeF((size_t)B * Hd);
    float* w2hB  = takeF((size_t)B * D2);
    float* outB  = takeF(BT);
    // packed f16 weights
    _Float16* pCtxWi[2] = { takeH((size_t)D3p16 * Dp),  takeH((size_t)D3p16 * Dp)  };
    _Float16* pCtxWh[2] = { takeH((size_t)D3p16 * Dp),  takeH((size_t)D3p16 * Dp)  };
    _Float16* pM0Wi[2]  = { takeH((size_t)D3p16 * D8),  takeH((size_t)D3p16 * D8)  };
    _Float16* pM0Wh[2]  = { takeH((size_t)D3p16 * Dp),  takeH((size_t)D3p16 * Dp)  };
    _Float16* pM1Wi[2]  = { takeH((size_t)D3p16 * D2p), takeH((size_t)D3p16 * D2p) };
    _Float16* pM1Wh[2]  = { takeH((size_t)D3p16 * Dp),  takeH((size_t)D3p16 * Dp)  };
    _Float16* pEncWi    = takeH((size_t)G12 * D2p);
    _Float16* pEncWh    = takeH((size_t)G12 * Hdp);
    _Float16* pW1       = takeH((size_t)208 * Hdp);
    _Float16* U16       = takeH(BL * D2p);           // [B][64][224]
    _Float16* Ut16      = takeH((size_t)B * 208 * L); // [B][208][64]

    float* GXf = GX;
    float* GXb = GX + BT * D3;
    float* QXf = GX;
    float* QXb = GX + BL * D3;

    // 0) zero padded-activation buffers (scans only write logical cols)
    { long n = BT * D2p; fill_zero<<<(n + 255) / 256, 256, 0, stream>>>(M0, n); }
    { long n = BT * D2p; fill_zero<<<(n + 255) / 256, 256, 0, stream>>>(M1, n); }
    { long n = BT * Hdp; fill_zero<<<(n + 255) / 256, 256, 0, stream>>>(encB, n); }

    // 1) pack weights to zero-padded f16
    for (int d = 0; d < 2; ++d) {
        packW(stream, ctx_wi + (long)d * D3 * D,   pCtxWi[d], D3, D,  D3p16, Dp);
        packW(stream, ctx_wh + (long)d * D3 * D,   pCtxWh[d], D3, D,  D3p16, Dp);
        packW(stream, mod0_wi + (long)d * D3 * D8, pM0Wi[d],  D3, D8, D3p16, D8);
        packW(stream, mod0_wh + (long)d * D3 * D,  pM0Wh[d],  D3, D,  D3p16, Dp);
        packW(stream, mod1_wi + (long)d * D3 * D2, pM1Wi[d],  D3, D2, D3p16, D2p);
        packW(stream, mod1_wh + (long)d * D3 * D,  pM1Wh[d],  D3, D,  D3p16, Dp);
    }
    packW(stream, enc_wi, pEncWi, G12, D2, G12, D2p);
    packW(stream, enc_wh, pEncWh, G12, Hd, G12, Hdp);
    packW(stream, w1,     pW1,    D2,  Hd, 208, Hdp);

    // 2) embedding gathers (padded)
    { long n = BT * Dp; gather_embed<<<(n + 255) / 256, 256, 0, stream>>>(x_word, embed, embC, D, Dp, n); }
    { long n = BL * Dp; gather_embed<<<(n + 255) / 256, 256, 0, stream>>>(x_query, embed, embQ, D, Dp, n); }

    // 3) context bi-GRU over words
    gemm(stream, embC, 0, Dp, pCtxWi[0], 0, Dp, ctx_bi,      GXf, 0, D3, (int)BT, D3, D3, Dp, 1);
    gemm(stream, embC, 0, Dp, pCtxWi[1], 0, Dp, ctx_bi + D3, GXb, 0, D3, (int)BT, D3, D3, Dp, 1);
    gru_scan(stream, GXf, pCtxWh[0], ctx_bh,      Hbuf, B, T, D, Dp, D2, 0, 0);
    gru_scan(stream, GXb, pCtxWh[1], ctx_bh + D3, Hbuf, B, T, D, Dp, D2, D, 1);

    // 4) context bi-GRU over query
    gemm(stream, embQ, 0, Dp, pCtxWi[0], 0, Dp, ctx_bi,      QXf, 0, D3, (int)BL, D3, D3, Dp, 1);
    gemm(stream, embQ, 0, Dp, pCtxWi[1], 0, Dp, ctx_bi + D3, QXb, 0, D3, (int)BL, D3, D3, Dp, 1);
    gru_scan(stream, QXf, pCtxWh[0], ctx_bh,      Ubuf, B, L, D, Dp, D2, 0, 0);
    gru_scan(stream, QXb, pCtxWh[1], ctx_bh + D3, Ubuf, B, L, D, Dp, D2, D, 1);

    // 5) trilinear similarity + attention
    rowdot<<<((int)BT + 127) / 128, 128, 0, stream>>>(Hbuf, w_sim,      hw1, (int)BT, D2);
    rowdot<<<((int)BL + 127) / 128, 128, 0, stream>>>(Ubuf, w_sim + D2, uw2, (int)BL, D2);
    { long n = BT * D2p; mul_cols_pad<<<(n + 255) / 256, 256, 0, stream>>>(Hbuf, w_sim + 2 * D2, Hw3, D2, D2p, n); }
    { long n = BL * D2p; pack_u16<<<(n + 255) / 256, 256, 0, stream>>>(Ubuf, U16, L, D2, D2p, n); }
    { long n = (long)B * 208 * L; pack_ut16<<<(n + 255) / 256, 256, 0, stream>>>(Ubuf, Ut16, D2, 208, L, n); }
    gemm(stream, Hw3, (long)T * D2p, D2p, U16, (long)L * D2p, D2p, nullptr,
         Sbuf, (long)T * L, L, T, L, L, D2p, B);
    softmax_S<<<(int)BT, 64, 0, stream>>>(Sbuf, hw1, uw2, SmaxB, T, L);
    softmax_T<<<B, 256, 0, stream>>>(SmaxB, Batt, T);
    q2c_kernel<<<B, 256, 0, stream>>>(Batt, Hbuf, q2cB, T, D2);
    gemm(stream, Sbuf, (long)T * L, L, Ut16, (long)208 * L, L, nullptr,
         c2qB, (long)T * D2, D2, T, D2, D2, L, B);
    { long n = BT * D8; build_G<<<(n + 255) / 256, 256, 0, stream>>>(Hbuf, c2qB, q2cB, Gbuf, B, T, D2); }

    // 6) modeling bi-GRU layer 0 (input 8D)
    gemm(stream, Gbuf, 0, D8, pM0Wi[0], 0, D8, mod0_bi,      GXf, 0, D3, (int)BT, D3, D3, D8, 1);
    gemm(stream, Gbuf, 0, D8, pM0Wi[1], 0, D8, mod0_bi + D3, GXb, 0, D3, (int)BT, D3, D3, D8, 1);
    gru_scan(stream, GXf, pM0Wh[0], mod0_bh,      M0, B, T, D, Dp, D2p, 0, 0);
    gru_scan(stream, GXb, pM0Wh[1], mod0_bh + D3, M0, B, T, D, Dp, D2p, D, 1);

    // 7) modeling bi-GRU layer 1 (input 2D)
    gemm(stream, M0, 0, D2p, pM1Wi[0], 0, D2p, mod1_bi,      GXf, 0, D3, (int)BT, D3, D3, D2p, 1);
    gemm(stream, M0, 0, D2p, pM1Wi[1], 0, D2p, mod1_bi + D3, GXb, 0, D3, (int)BT, D3, D3, D2p, 1);
    gru_scan(stream, GXf, pM1Wh[0], mod1_bh,      M1, B, T, D, Dp, D2p, 0, 0);
    gru_scan(stream, GXb, pM1Wh[1], mod1_bh + D3, M1, B, T, D, Dp, D2p, D, 1);

    // 8) pointer encoder GRU (hidden 4D) + W1 projection
    gemm(stream, M1, 0, D2p, pEncWi, 0, D2p, enc_bi, GX, 0, G12, (int)BT, G12, G12, D2p, 1);
    gru_scan(stream, GX, pEncWh, enc_bh, encB, B, T, Hd, Hdp, Hdp, 0, 0);
    gemm(stream, encB, 0, Hdp, pW1, 0, Hdp, nullptr, encW1, 0, D2, (int)BT, D2, D2, Hdp, 1);

    // 9) decoder: two pointer steps
    size_t decSh = (size_t)(2 * B * Hd) * sizeof(float);
    for (int s = 0; s < 2; ++s) {
        dec_cell<<<1, 512, decSh, stream>>>(hid, s == 0 ? 1 : 0, dec_wh, dec_bi, dec_bh,
                                            w2, hid, w2hB, B, Hd, D2);
        blend_kernel<<<((int)BT + 255) / 256, 256, 0, stream>>>(encW1, w2hB, vt, outB, B, T, D2);
        logsoftmax_T<<<B, 256, 0, stream>>>(outB, (float*)d_out, T, s, 2);
    }
}
